// RGLRU_52321291600330
// MI455X (gfx1250) — compile-verified, matrix operand order checked by reading, not run on hardware
//
#include <hip/hip_runtime.h>
#include <math.h>

// ---------------------------------------------------------------------------
// RG-LRU forward for MI455X (gfx1250, wave32, WMMA 16x16x32 bf16->f32).
// Tile staging uses CDNA5 async Global->LDS copies (ASYNCcnt) when available.
// ---------------------------------------------------------------------------
typedef __attribute__((ext_vector_type(16))) __bf16 v16bf;
typedef __attribute__((ext_vector_type(8)))  float  v8f;
typedef __attribute__((__vector_size__(16))) int    v4i;   // builtin arg type

#if __has_builtin(__builtin_amdgcn_global_load_async_to_lds_b128)
#  define HAVE_ASYNC_LDS 1
#else
#  define HAVE_ASYNC_LDS 0
#endif

__device__ __forceinline__ void async_wait0() {
#if __has_builtin(__builtin_amdgcn_s_wait_asynccnt)
    __builtin_amdgcn_s_wait_asynccnt(0);
#else
    asm volatile("s_wait_asynccnt 0x0" ::: "memory");
#endif
}

union BF16Frag { v16bf v; unsigned int d[8]; };

__device__ __forceinline__ unsigned short f2bf(float f) {
    unsigned int u = __float_as_uint(f);
    u += 0x7FFFu + ((u >> 16) & 1u);     // round-to-nearest-even
    return (unsigned short)(u >> 16);
}
__device__ __forceinline__ unsigned int pack2bf(float lo, float hi) {
    return (unsigned int)f2bf(lo) | ((unsigned int)f2bf(hi) << 16);
}

struct alignas(8) U32x2 { unsigned int x, y; };

#define LDA_S 40   // padded LDS row stride (ushorts): 80B rows, 16B aligned
#define LDB_S 40

// ---------------------------------------------------------------------------
// Elementwise f32 -> bf16 (4 elements/thread, b128 in / b64 out)
// ---------------------------------------------------------------------------
__global__ __launch_bounds__(256)
void convert_bf16(const float* __restrict__ in, unsigned short* __restrict__ outp) {
    size_t i = (size_t)blockIdx.x * 256 + threadIdx.x;   // float4 units
    float4 f = ((const float4*)in)[i];
    U32x2 p = { pack2bf(f.x, f.y), pack2bf(f.z, f.w) };
    *(U32x2*)&outp[i * 4] = p;
}

// ---------------------------------------------------------------------------
// Tiled transpose + convert: Wt[z][n*K + k] = bf16(W[z][k*N + n])
// ---------------------------------------------------------------------------
__global__ __launch_bounds__(256)
void transpose_convert(const float* __restrict__ W, unsigned short* __restrict__ Wt,
                       int K, int N,
                       unsigned long long inStride, unsigned long long outStride) {
    __shared__ float t[32][33];
    const float* Wz = W + (size_t)blockIdx.z * inStride;
    unsigned short* Wtz = Wt + (size_t)blockIdx.z * outStride;
    const int kt = blockIdx.x * 32, nt = blockIdx.y * 32;
    const int tx = threadIdx.x & 31;          // n offset on read
    const int ty = threadIdx.x >> 5;          // 0..7
    #pragma unroll
    for (int j = 0; j < 4; ++j) {
        int k = ty + 8 * j;
        t[k][tx] = Wz[(size_t)(kt + k) * N + (nt + tx)];
    }
    __syncthreads();
    #pragma unroll
    for (int i = 0; i < 2; ++i) {
        int u  = i * 256 + threadIdx.x;       // 512 units: 32 n x 16 k-pairs
        int n  = u >> 4;
        int ku = (u & 15) * 2;
        unsigned int v = pack2bf(t[ku][n], t[ku + 1][n]);
        *(unsigned int*)&Wtz[(size_t)(nt + n) * K + (kt + ku)] = v;
    }
}

// ---------------------------------------------------------------------------
// C[M,N] = A[M,K] * Bt[N,K]^T + bias[N]   (bf16 inputs, f32 out, optional
// bf16 shadow copy Cb). block = 128 threads (4 waves), tile 128x64.
// ---------------------------------------------------------------------------
__global__ __launch_bounds__(128)
void wmma_gemm_bias(const unsigned short* __restrict__ A,
                    const unsigned short* __restrict__ Bt,
                    const float* __restrict__ bias, float* __restrict__ C,
                    unsigned short* __restrict__ Cb,
                    int M, int N, int K) {
    __shared__ unsigned short ldsA[128 * LDA_S];
    __shared__ unsigned short ldsB[64  * LDB_S];

    const int tid   = threadIdx.x;
    const int lane  = tid & 31;
    const int wave  = tid >> 5;
    const int laneM = lane & 15;
    const int kb    = (lane >> 4) * 8;       // K-group base per lane half
    const int n0    = blockIdx.x * 64;
    const int m0    = blockIdx.y * 128;

    v8f zero = {};
    v8f acc[2][4];
    for (int i = 0; i < 2; ++i)
        for (int j = 0; j < 4; ++j) acc[i][j] = zero;

    for (int k0 = 0; k0 < K; k0 += 32) {
#if HAVE_ASYNC_LDS
        // async Global->LDS b128 copies (ASYNCcnt), bypassing VGPRs
        #pragma unroll
        for (int i = 0; i < 4; ++i) {
            int u = i * 128 + tid;
            int r = u >> 2, k = (u & 3) * 8;
            __builtin_amdgcn_global_load_async_to_lds_b128(
                (v4i*)(A + (size_t)(m0 + r) * K + (k0 + k)),
                (v4i*)&ldsA[r * LDA_S + k], 0, 0);
        }
        #pragma unroll
        for (int i = 0; i < 2; ++i) {
            int u = i * 128 + tid;
            int n = u >> 2, k = (u & 3) * 8;
            __builtin_amdgcn_global_load_async_to_lds_b128(
                (v4i*)(Bt + (size_t)(n0 + n) * K + (k0 + k)),
                (v4i*)&ldsB[n * LDB_S + k], 0, 0);
        }
        async_wait0();
#else
        // fallback: distinct temporaries so loads pipeline before stores
        uint4 tA[4], tB[2];
        #pragma unroll
        for (int i = 0; i < 4; ++i) {
            int u = i * 128 + tid;
            int r = u >> 2, k = (u & 3) * 8;
            tA[i] = *(const uint4*)(A + (size_t)(m0 + r) * K + (k0 + k));
        }
        #pragma unroll
        for (int i = 0; i < 2; ++i) {
            int u = i * 128 + tid;
            int n = u >> 2, k = (u & 3) * 8;
            tB[i] = *(const uint4*)(Bt + (size_t)(n0 + n) * K + (k0 + k));
        }
        #pragma unroll
        for (int i = 0; i < 4; ++i) {
            int u = i * 128 + tid;
            int r = u >> 2, k = (u & 3) * 8;
            *(uint4*)&ldsA[r * LDA_S + k] = tA[i];
        }
        #pragma unroll
        for (int i = 0; i < 2; ++i) {
            int u = i * 128 + tid;
            int n = u >> 2, k = (u & 3) * 8;
            *(uint4*)&ldsB[n * LDB_S + k] = tB[i];
        }
#endif
        if (k0 + 32 < K)   // global_prefetch_b8 for next A tile
            __builtin_prefetch(A + (size_t)(m0 + (tid & 127)) * K + (k0 + 32));
        __syncthreads();

        BF16Frag aF[2], bF[4];
        #pragma unroll
        for (int rm = 0; rm < 2; ++rm) {
            int row = wave * 32 + rm * 16 + laneM;
            const unsigned short* p = &ldsA[row * LDA_S];
            #pragma unroll
            for (int r = 0; r < 8; ++r) {
                int k = kb + (r & 3) * 2 + (r >> 2) * 16;
                aF[rm].d[r] = *(const unsigned int*)(p + k);
            }
        }
        #pragma unroll
        for (int cn = 0; cn < 4; ++cn) {
            int col = cn * 16 + laneM;
            const unsigned short* p = &ldsB[col * LDB_S];
            #pragma unroll
            for (int r = 0; r < 8; ++r) {
                int k = kb + (r & 3) * 2 + (r >> 2) * 16;
                bF[cn].d[r] = *(const unsigned int*)(p + k);
            }
        }
        #pragma unroll
        for (int rm = 0; rm < 2; ++rm)
            #pragma unroll
            for (int cn = 0; cn < 4; ++cn)
                acc[rm][cn] = __builtin_amdgcn_wmma_f32_16x16x32_bf16(
                    false, aF[rm].v, false, bF[cn].v,
                    (short)0, acc[rm][cn], false, false);
        __syncthreads();
    }

    // epilogue: D layout (lane<16: M=v, N=lane ; lane>=16: M=v+8, N=lane-16)
    const int rowBase = m0 + wave * 32;
    const int mh = 8 * (lane >> 4);
    #pragma unroll
    for (int rm = 0; rm < 2; ++rm) {
        #pragma unroll
        for (int cn = 0; cn < 4; ++cn) {
            int col = n0 + cn * 16 + laneM;
            float bv = bias[col];
            #pragma unroll
            for (int v = 0; v < 8; ++v) {
                int row = rowBase + rm * 16 + v + mh;
                float val = acc[rm][cn][v] + bv;
                C[(size_t)row * N + col] = val;
                if (Cb) Cb[(size_t)row * N + col] = f2bf(val);
            }
        }
    }
}

// ---------------------------------------------------------------------------
// Fused block-diagonal gate GEMMs + RG-LRU coefficient epilogue.
// ---------------------------------------------------------------------------
__global__ __launch_bounds__(128)
void gates_kernel(const float* __restrict__ xh,
                  const unsigned short* __restrict__ xhb,
                  const unsigned short* __restrict__ WgxT, const float* __restrict__ bgx,
                  const unsigned short* __restrict__ WgaT, const float* __restrict__ bga,
                  const float* __restrict__ aRealP, const float* __restrict__ aImagP,
                  float4* __restrict__ coef) {
    __shared__ unsigned short ldsA [64  * LDA_S];
    __shared__ unsigned short ldsBX[128 * LDB_S];
    __shared__ unsigned short ldsBA[128 * LDB_S];

    const int tid   = threadIdx.x;
    const int lane  = tid & 31;
    const int wave  = tid >> 5;
    const int laneM = lane & 15;
    const int kb    = (lane >> 4) * 8;
    const int nblk  = blockIdx.x;          // 0..7
    const int m0    = blockIdx.y * 64;

    const unsigned short* Wx = WgxT + (size_t)nblk * 128 * 256;  // [n][k]
    const unsigned short* Wa = WgaT + (size_t)nblk * 128 * 256;

    v8f zero = {};
    v8f accX[8], accA[8];
    for (int i = 0; i < 8; ++i) { accX[i] = zero; accA[i] = zero; }

    for (int k0 = 0; k0 < 256; k0 += 32) {
#if HAVE_ASYNC_LDS
        #pragma unroll
        for (int i = 0; i < 2; ++i) {
            int u = i * 128 + tid;
            int r = u >> 2, k = (u & 3) * 8;
            __builtin_amdgcn_global_load_async_to_lds_b128(
                (v4i*)(xhb + (size_t)(m0 + r) * 2048 + nblk * 256 + k0 + k),
                (v4i*)&ldsA[r * LDA_S + k], 0, 0);
        }
        #pragma unroll
        for (int i = 0; i < 4; ++i) {
            int u = i * 128 + tid;
            int n = u >> 2, k = (u & 3) * 8;
            __builtin_amdgcn_global_load_async_to_lds_b128(
                (v4i*)(Wx + (size_t)n * 256 + (k0 + k)),
                (v4i*)&ldsBX[n * LDB_S + k], 0, 0);
            __builtin_amdgcn_global_load_async_to_lds_b128(
                (v4i*)(Wa + (size_t)n * 256 + (k0 + k)),
                (v4i*)&ldsBA[n * LDB_S + k], 0, 0);
        }
        async_wait0();
#else
        uint4 tA[2], tX[4], tW[4];
        #pragma unroll
        for (int i = 0; i < 2; ++i) {
            int u = i * 128 + tid;
            int r = u >> 2, k = (u & 3) * 8;
            tA[i] = *(const uint4*)(xhb + (size_t)(m0 + r) * 2048 +
                                    nblk * 256 + k0 + k);
        }
        #pragma unroll
        for (int i = 0; i < 4; ++i) {
            int u = i * 128 + tid;
            int n = u >> 2, k = (u & 3) * 8;
            tX[i] = *(const uint4*)(Wx + (size_t)n * 256 + (k0 + k));
            tW[i] = *(const uint4*)(Wa + (size_t)n * 256 + (k0 + k));
        }
        #pragma unroll
        for (int i = 0; i < 2; ++i) {
            int u = i * 128 + tid;
            int r = u >> 2, k = (u & 3) * 8;
            *(uint4*)&ldsA[r * LDA_S + k] = tA[i];
        }
        #pragma unroll
        for (int i = 0; i < 4; ++i) {
            int u = i * 128 + tid;
            int n = u >> 2, k = (u & 3) * 8;
            *(uint4*)&ldsBX[n * LDB_S + k] = tX[i];
            *(uint4*)&ldsBA[n * LDB_S + k] = tW[i];
        }
#endif
        __syncthreads();

        BF16Frag aF;
        {
            int row = wave * 16 + laneM;
            const unsigned short* p = &ldsA[row * LDA_S];
            #pragma unroll
            for (int r = 0; r < 8; ++r) {
                int k = kb + (r & 3) * 2 + (r >> 2) * 16;
                aF.d[r] = *(const unsigned int*)(p + k);
            }
        }
        #pragma unroll
        for (int cn = 0; cn < 8; ++cn) {
            int col = cn * 16 + laneM;
            BF16Frag bX, bA;
            const unsigned short* px = &ldsBX[col * LDB_S];
            const unsigned short* pa = &ldsBA[col * LDB_S];
            #pragma unroll
            for (int r = 0; r < 8; ++r) {
                int k = kb + (r & 3) * 2 + (r >> 2) * 16;
                bX.d[r] = *(const unsigned int*)(px + k);
                bA.d[r] = *(const unsigned int*)(pa + k);
            }
            accX[cn] = __builtin_amdgcn_wmma_f32_16x16x32_bf16(
                false, aF.v, false, bX.v, (short)0, accX[cn], false, false);
            accA[cn] = __builtin_amdgcn_wmma_f32_16x16x32_bf16(
                false, aF.v, false, bA.v, (short)0, accA[cn], false, false);
        }
        __syncthreads();
    }

    // epilogue: sigmoid gates -> complex decay + scan inputs (one b128 store)
    const int mh = 8 * (lane >> 4);
    #pragma unroll
    for (int cn = 0; cn < 8; ++cn) {
        int chan = nblk * 128 + cn * 16 + laneM;         // 0..1023
        float sp  = log1pf(expf(aRealP[chan]));          // softplus
        float aip = aImagP[chan];
        float bx  = bgx[chan], ba = bga[chan];
        #pragma unroll
        for (int v = 0; v < 8; ++v) {
            int row  = m0 + wave * 16 + v + mh;
            float gx = 1.0f / (1.0f + expf(-(accX[cn][v] + bx)));
            float ga = 1.0f / (1.0f + expf(-(accA[cn][v] + ba)));
            float log_a = -8.0f * ga * sp;
            float mag   = expf(log_a);
            float th    = aip * ga;
            float sn, cs;
            sincosf(th, &sn, &cs);
            float a_sq = mag * mag;                      // exp(2*log_a)
            float norm = sqrtf(fmaxf(1.0f - a_sq, 0.0f));
            float g    = norm * gx;
            coef[(size_t)row * 1024 + chan] = make_float4(
                mag * cs, mag * sn,
                g * xh[(size_t)row * 2048 + chan],
                g * xh[(size_t)row * 2048 + 1024 + chan]);
        }
    }
}

// ---------------------------------------------------------------------------
// Chunked complex scan: S=4096 split into 32 chunks of 128.
// ---------------------------------------------------------------------------
__global__ __launch_bounds__(256)
void scan_pass1(const float4* __restrict__ coef, float4* __restrict__ carry) {
    int tid = blockIdx.x * 256 + threadIdx.x;    // B*32*1024 = 131072
    int c = tid & 1023;
    int j = (tid >> 10) & 31;
    int b = tid >> 15;
    float Ar = 1.0f, Ai = 0.0f, Xr = 0.0f, Xi = 0.0f;
    size_t base = ((size_t)b * 4096 + (size_t)j * 128) * 1024 + c;
    for (int i = 0; i < 128; ++i) {
        float4 q = coef[base + (size_t)i * 1024];
        float nAr = q.x * Ar - q.y * Ai;
        float nAi = q.x * Ai + q.y * Ar;
        float nXr = q.x * Xr - q.y * Xi + q.z;
        float nXi = q.x * Xi + q.y * Xr + q.w;
        Ar = nAr; Ai = nAi; Xr = nXr; Xi = nXi;
    }
    carry[((size_t)b * 1024 + c) * 32 + j] = make_float4(Ar, Ai, Xr, Xi);
}

__global__ __launch_bounds__(256)
void scan_pass2(const float4* __restrict__ carry, float2* __restrict__ pref,
                float* __restrict__ hlast) {
    int tid = blockIdx.x * 256 + threadIdx.x;    // 4096
    int c = tid & 1023;
    int b = tid >> 10;
    float hr = 0.0f, hi = 0.0f;
    size_t base = ((size_t)b * 1024 + c) * 32;
    for (int j = 0; j < 32; ++j) {
        pref[base + j] = make_float2(hr, hi);    // exclusive prefix
        float4 cv = carry[base + j];
        float nr = cv.x * hr - cv.y * hi + cv.z;
        float ni = cv.x * hi + cv.y * hr + cv.w;
        hr = nr; hi = ni;
    }
    hlast[(size_t)b * 2048 + c]        = hr;     // concat([hr_last, hi_last])
    hlast[(size_t)b * 2048 + 1024 + c] = hi;
}

__global__ __launch_bounds__(256)
void scan_pass3(const float4* __restrict__ coef, const float2* __restrict__ pref,
                unsigned short* __restrict__ hb) {
    int tid = blockIdx.x * 256 + threadIdx.x;
    int c = tid & 1023;
    int j = (tid >> 10) & 31;
    int b = tid >> 15;
    float2 p = pref[((size_t)b * 1024 + c) * 32 + j];
    float hr = p.x, hi = p.y;
    size_t base = ((size_t)b * 4096 + (size_t)j * 128) * 1024 + c;
    for (int i = 0; i < 128; ++i) {
        float4 q = coef[base + (size_t)i * 1024];
        float nr = q.x * hr - q.y * hi + q.z;
        float ni = q.x * hi + q.y * hr + q.w;
        hr = nr; hi = ni;
        size_t s = (size_t)b * 4096 + (size_t)j * 128 + i;
        hb[s * 2048 + c]        = f2bf(hr);      // bf16 h for GEMM2
        hb[s * 2048 + 1024 + c] = f2bf(hi);
    }
}

// ---------------------------------------------------------------------------
extern "C" void kernel_launch(void* const* d_in, const int* in_sizes, int n_in,
                              void* d_out, int out_size, void* d_ws, size_t ws_size,
                              hipStream_t stream) {
    const float* x    = (const float*)d_in[0];
    const float* aRp  = (const float*)d_in[1];
    const float* aIp  = (const float*)d_in[2];
    const float* Win  = (const float*)d_in[3];
    const float* bin  = (const float*)d_in[4];
    const float* Wgx  = (const float*)d_in[5];
    const float* bgx  = (const float*)d_in[6];
    const float* Wga  = (const float*)d_in[7];
    const float* bga  = (const float*)d_in[8];
    const float* Wout = (const float*)d_in[9];
    const float* bout = (const float*)d_in[10];
    float* out = (float*)d_out;

    // ---- workspace carve-up (byte offsets, all 16B aligned) ----
    char* w = (char*)d_ws;
    float*          xh    = (float*)          (w);                // 128 MB f32
    unsigned short* hb    = (unsigned short*) (w);                // aliases xh
    float4*         coef  = (float4*)         (w + 134217728ull); // 256 MB
    float4*         carry = (float4*)         (w + 402653184ull); //   2 MB
    float2*         pref  = (float2*)         (w + 404750336ull); //   1 MB
    unsigned short* xb    = (unsigned short*) (w + 405798912ull); //  64 MB
    unsigned short* xhb   = (unsigned short*) (w + 472907776ull); //  64 MB
    unsigned short* WinT  = (unsigned short*) (w + 540016640ull); //   8 MB
    unsigned short* WoutT = (unsigned short*) (w + 548405248ull); //   8 MB
    unsigned short* WgxT  = (unsigned short*) (w + 556793856ull); // 0.5 MB
    unsigned short* WgaT  = (unsigned short*) (w + 557318144ull); // 0.5 MB

    // ---- 0) one-time conversions / transposes ----
    convert_bf16<<<32768, 256, 0, stream>>>(x, xb);      // 8M float4 units
    transpose_convert<<<dim3(64, 64, 1), 256, 0, stream>>>(
        Win, WinT, 2048, 2048, 0ull, 0ull);
    transpose_convert<<<dim3(64, 64, 1), 256, 0, stream>>>(
        Wout, WoutT, 2048, 2048, 0ull, 0ull);
    transpose_convert<<<dim3(8, 4, 8), 256, 0, stream>>>(
        Wgx, WgxT, 256, 128, 32768ull, 32768ull);
    transpose_convert<<<dim3(8, 4, 8), 256, 0, stream>>>(
        Wga, WgaT, 256, 128, 32768ull, 32768ull);

    dim3 blk(128);
    // 1) x_h = x @ W_in + b_in   (f32 + bf16 shadow)
    wmma_gemm_bias<<<dim3(2048 / 64, 16384 / 128), blk, 0, stream>>>(
        xb, WinT, bin, xh, xhb, 16384, 2048, 2048);
    // 2) gates + interleaved scan coefficients
    gates_kernel<<<dim3(8, 16384 / 64), blk, 0, stream>>>(
        xh, xhb, WgxT, bgx, WgaT, bga, aRp, aIp, coef);
    // 3) chunked complex scan (pass3 emits bf16 h, aliasing dead x_h)
    scan_pass1<<<512, 256, 0, stream>>>(coef, carry);
    scan_pass2<<<16, 256, 0, stream>>>(carry, pref, out + 33554432);
    scan_pass3<<<512, 256, 0, stream>>>(coef, pref, hb);
    // 4) y = h @ W_out + b_out  (written straight into d_out)
    wmma_gemm_bias<<<dim3(2048 / 64, 16384 / 128), blk, 0, stream>>>(
        hb, WoutT, bout, out, (unsigned short*)nullptr, 16384, 2048, 2048);
}
